// HeteroGNN_19155554140463
// MI455X (gfx1250) — compile-verified, matrix operand order checked by reading, not run on hardware
//
#include <hip/hip_runtime.h>
#include <math.h>

// Problem constants (match reference)
#define NE_    100000
#define NPER_  50000
#define DEMP_  64
#define DPER_  32
#define H_     128
#define OUT_   64
#define EEE_   1600000
#define EPE_   800000
#define EEP_   800000

typedef __attribute__((ext_vector_type(16))) __bf16 v16bf;
typedef __attribute__((ext_vector_type(8)))  float  v8f;

// ---------------------------------------------------------------------------
// WMMA GEMM: Y[M,N] = op( X[M,K] @ W[K,N] + bias (+ Y if accumulate) )
// X,W,Y f32 in global; operands converted to bf16 (hw v_cvt, RNE), f32
// accumulate via v_wmma_f32_16x16x32_bf16.
//
// No LDS, no barriers: each lane loads its A fragment directly as four
// global_load_b128 (two contiguous 8-float runs of one row, per the ISA
// 7.12.2 16-bit A layout) and its B fragment as 16 strided b32 loads of the
// (L2-resident) weight matrix.  OOB M rows are index-clamped, never stored,
// so EXEC stays all-ones for WMMA and no branch diamonds are emitted.
//
// Each wave computes a 16x32 output strip: 2 N-tiles reuse one A fragment.
// Requires: N % 32 == 0, K % 32 == 0 (all call sites: N in {64,128},
// K in {32,64,128}).
// ---------------------------------------------------------------------------
#define GEMM_WAVES 8
__global__ void gemm_bf16_wmma(const float* __restrict__ X,
                               const float* __restrict__ W,
                               const float* __restrict__ bias,
                               float* __restrict__ Y,
                               int M, int N, int K, int accumulate, int relu_act) {
  const int lane = threadIdx.x & 31;
  const int wv   = threadIdx.x >> 5;
  const int n0   = blockIdx.x * 32;                       // 2 x 16 cols
  const int m0   = (blockIdx.y * GEMM_WAVES + wv) * 16;   // 16 rows

  const int p  = lane & 15;            // M index for A, N index for B
  const int kb = (lane >> 4) << 3;     // 0 or 8 (K sub-block)

  const int  rowA = min(m0 + p, M - 1);          // clamp: EXEC stays full
  const bool rowsValid = (m0 + p) < M;           // per-lane store guard only
  (void)rowsValid;
  const float* __restrict__ arow  = X + (size_t)rowA * K;
  const float* __restrict__ bcol0 = W + (n0 + p);
  const float* __restrict__ bcol1 = W + (n0 + 16 + p);

  v8f acc0 = {0.f, 0.f, 0.f, 0.f, 0.f, 0.f, 0.f, 0.f};
  v8f acc1 = {0.f, 0.f, 0.f, 0.f, 0.f, 0.f, 0.f, 0.f};

  for (int k0 = 0; k0 < K; k0 += 32) {
    // ---- A fragment: two contiguous 8-float runs of this lane's row ----
    const float4 a0 = *(const float4*)(arow + k0 + kb + 0);
    const float4 a1 = *(const float4*)(arow + k0 + kb + 4);
    const float4 a2 = *(const float4*)(arow + k0 + kb + 16);
    const float4 a3 = *(const float4*)(arow + k0 + kb + 20);
    if (k0 + 32 < K) __builtin_prefetch(arow + k0 + 32, 0, 1);

    v16bf af;
    af[0]  = (__bf16)a0.x; af[1]  = (__bf16)a0.y; af[2]  = (__bf16)a0.z; af[3]  = (__bf16)a0.w;
    af[4]  = (__bf16)a1.x; af[5]  = (__bf16)a1.y; af[6]  = (__bf16)a1.z; af[7]  = (__bf16)a1.w;
    af[8]  = (__bf16)a2.x; af[9]  = (__bf16)a2.y; af[10] = (__bf16)a2.z; af[11] = (__bf16)a2.w;
    af[12] = (__bf16)a3.x; af[13] = (__bf16)a3.y; af[14] = (__bf16)a3.z; af[15] = (__bf16)a3.w;

    // ---- B fragments: strided column reads (weights are cache-hot) ----
    v16bf bf0, bf1;
#pragma unroll
    for (int e = 0; e < 16; ++e) {
      const int kk = kb + e + ((e & 8) ? 8 : 0);          // 0..31 within tile
      const size_t off = (size_t)(k0 + kk) * N;
      bf0[e] = (__bf16)bcol0[off];
      bf1[e] = (__bf16)bcol1[off];
    }

    acc0 = __builtin_amdgcn_wmma_f32_16x16x32_bf16(
        false, af, false, bf0, (short)0, acc0, false, false);
    acc1 = __builtin_amdgcn_wmma_f32_16x16x32_bf16(
        false, af, false, bf1, (short)0, acc1, false, false);
  }

  // C/D layout: lane l -> col (l&15); VGPR v -> row m0 + v + (l>=16 ? 8 : 0)
  const int rbase = m0 + ((lane >> 4) << 3);
#pragma unroll
  for (int v = 0; v < 8; ++v) {
    const int row = rbase + v;
    if (row < M) {
      const int c0 = n0 + p, c1 = n0 + 16 + p;
      size_t o0 = (size_t)row * N + c0;
      size_t o1 = (size_t)row * N + c1;
      float v0 = acc0[v], v1 = acc1[v];
      if (bias)       { v0 += bias[c0]; v1 += bias[c1]; }
      if (accumulate) { v0 += Y[o0];    v1 += Y[o1]; }
      if (relu_act)   { v0 = fmaxf(v0, 0.f); v1 = fmaxf(v1, 0.f); }
      Y[o0] = v0;
      Y[o1] = v1;
    }
  }
}

// ---------------------------------------------------------------------------
// Elementwise / edge kernels
// ---------------------------------------------------------------------------
__global__ void fill_val(float* p, float v, long long n) {
  long long i = (long long)blockIdx.x * blockDim.x + threadIdx.x;
  if (i < n) p[i] = v;
}

__global__ void fill_bias_bcast(float* p, const float* __restrict__ b,
                                long long n, int F) {
  long long i = (long long)blockIdx.x * blockDim.x + threadIdx.x;
  if (i < n) p[i] = b[(int)(i % F)];
}

// agg[dst] += h_src[src]  (float4 vectorized) ; cnt[dst] += 1 on chunk 0
__global__ void edge_gather_add(const float* __restrict__ hsrc,
                                const int* __restrict__ src,
                                const int* __restrict__ dst,
                                float* agg, float* cnt, int E, int F) {
  long long t = (long long)blockIdx.x * blockDim.x + threadIdx.x;
  const int ch = F >> 2;
  long long total = (long long)E * ch;
  if (t >= total) return;
  int e = (int)(t / ch);
  int c = (int)(t % ch) * 4;
  int s = src[e], d = dst[e];
  const float4 v = *(const float4*)(hsrc + (size_t)s * F + c);
  float* p = agg + (size_t)d * F + c;
  atomicAdd(p + 0, v.x);
  atomicAdd(p + 1, v.y);
  atomicAdd(p + 2, v.z);
  atomicAdd(p + 3, v.w);
  if (c == 0) atomicAdd(cnt + d, 1.0f);
}

__global__ void mean_div(float* agg, const float* __restrict__ cnt,
                         long long total, int F) {
  long long t = (long long)blockIdx.x * blockDim.x + threadIdx.x;
  if (t >= total) return;
  int node = (int)(t / F);
  agg[t] *= (1.0f / fmaxf(cnt[node], 1.0f));
}

__global__ void matvec_dot(const float* __restrict__ Z,
                           const float* __restrict__ a,
                           float* out, int n, int F) {
  int i = blockIdx.x * blockDim.x + threadIdx.x;
  if (i >= n) return;
  const float* z = Z + (size_t)i * F;
  float s = 0.f;
#pragma unroll 8
  for (int f = 0; f < F; ++f) s += z[f] * a[f];
  out[i] = s;
}

__device__ __forceinline__ void atomicMaxF(float* addr, float val) {
  if (val >= 0.f) atomicMax((int*)addr, __float_as_int(val));
  else            atomicMin((unsigned int*)addr, __float_as_uint(val));
}

__global__ void gat_logit_max(const float* __restrict__ es,
                              const float* __restrict__ ed,
                              const int* __restrict__ src,
                              const int* __restrict__ dst,
                              float* logit, float* mmax, int E) {
  int i = blockIdx.x * blockDim.x + threadIdx.x;
  if (i >= E) return;
  float e = es[src[i]] + ed[dst[i]];
  e = (e > 0.f) ? e : 0.2f * e;        // leaky_relu(0.2)
  logit[i] = e;
  atomicMaxF(&mmax[dst[i]], e);
}

__global__ void gat_exp_den(float* logit, const float* __restrict__ mmax,
                            const int* __restrict__ dst, float* den, int E) {
  int i = blockIdx.x * blockDim.x + threadIdx.x;
  if (i >= E) return;
  float ex = __expf(logit[i] - mmax[dst[i]]);
  logit[i] = ex;
  atomicAdd(&den[dst[i]], ex);
}

__global__ void gat_scatter(const float* __restrict__ exv,
                            const float* __restrict__ den,
                            const float* __restrict__ zs,
                            const int* __restrict__ src,
                            const int* __restrict__ dst,
                            float* out, int E, int F) {
  long long t = (long long)blockIdx.x * blockDim.x + threadIdx.x;
  const int ch = F >> 2;
  long long total = (long long)E * ch;
  if (t >= total) return;
  int e = (int)(t / ch);
  int c = (int)(t % ch) * 4;
  float alpha = exv[e] / den[dst[e]];
  const float4 v = *(const float4*)(zs + (size_t)src[e] * F + c);
  float* p = out + (size_t)dst[e] * F + c;
  atomicAdd(p + 0, v.x * alpha);
  atomicAdd(p + 1, v.y * alpha);
  atomicAdd(p + 2, v.z * alpha);
  atomicAdd(p + 3, v.w * alpha);
}

// ---------------------------------------------------------------------------
static inline unsigned nblk(long long n, int b) { return (unsigned)((n + b - 1) / b); }

extern "C" void kernel_launch(void* const* d_in, const int* in_sizes, int n_in,
                              void* d_out, int out_size, void* d_ws, size_t ws_size,
                              hipStream_t stream) {
  (void)in_sizes; (void)n_in; (void)out_size; (void)ws_size;

  const float* x_emp   = (const float*)d_in[0];
  const float* x_per   = (const float*)d_in[1];
  const int*   ee_src  = (const int*)d_in[2];
  const int*   ee_dst  = (const int*)d_in[3];
  const int*   pe_src  = (const int*)d_in[4];
  const int*   pe_dst  = (const int*)d_in[5];
  const int*   ep_src  = (const int*)d_in[6];
  const int*   ep_dst  = (const int*)d_in[7];
  const float* lin_emp_w = (const float*)d_in[8];
  const float* lin_emp_b = (const float*)d_in[9];
  const float* lin_per_w = (const float*)d_in[10];
  const float* lin_per_b = (const float*)d_in[11];
  const float* s1_ee_Wl = (const float*)d_in[12];
  const float* s1_ee_bl = (const float*)d_in[13];
  const float* s1_ee_Wr = (const float*)d_in[14];
  const float* s1_pe_Wl = (const float*)d_in[15];
  const float* s1_pe_bl = (const float*)d_in[16];
  const float* s1_pe_Wr = (const float*)d_in[17];
  const float* s1_ep_Wl = (const float*)d_in[18];
  const float* s1_ep_bl = (const float*)d_in[19];
  const float* s1_ep_Wr = (const float*)d_in[20];
  const float* gat_Wsrc = (const float*)d_in[21];
  const float* gat_Wdst = (const float*)d_in[22];
  const float* gat_asrc = (const float*)d_in[23];
  const float* gat_adst = (const float*)d_in[24];
  const float* gat_b    = (const float*)d_in[25];
  const float* s2_pe_Wl = (const float*)d_in[26];
  const float* s2_pe_bl = (const float*)d_in[27];
  const float* s2_pe_Wr = (const float*)d_in[28];
  const float* s2_ep_Wl = (const float*)d_in[29];
  const float* s2_ep_bl = (const float*)d_in[30];
  const float* s2_ep_Wr = (const float*)d_in[31];

  // ---- workspace layout (floats), ~205 MB ----
  float* ws = (float*)d_ws;
  size_t o = 0;
  float* h_emp = ws + o; o += (size_t)NE_ * H_;
  float* h_per = ws + o; o += (size_t)NPER_ * H_;
  float* emp1  = ws + o; o += (size_t)NE_ * H_;
  float* per1  = ws + o; o += (size_t)NPER_ * H_;
  float* agg   = ws + o; o += (size_t)NE_ * H_;   // reused per relation
  float* cnt   = ws + o; o += (size_t)NE_;        // reused per relation
  // aliases into dead regions (h_emp/h_per dead after conv1):
  float* zs    = h_emp;                              // NE*OUT
  float* zd    = h_emp + (size_t)NE_ * OUT_;         // NE*OUT (fits in NE*H)
  float* logit = h_per;                              // E_EE (1.6M <= NPER*H)
  float* es    = h_per + (size_t)EEE_;
  float* ed    = es + NE_;
  float* mmax  = ed + NE_;
  float* den   = mmax + NE_;

  float* out_emp = (float*)d_out;                    // [NE, OUT]
  float* out_per = out_emp + (size_t)NE_ * OUT_;     // [NPER, OUT]

  auto GEMM = [&](const float* X, const float* W, const float* b, float* Y,
                  int M, int N, int K, int accum, int relu) {
    dim3 grid(N / 32, nblk(M, 16 * GEMM_WAVES));
    gemm_bf16_wmma<<<grid, 32 * GEMM_WAVES, 0, stream>>>(X, W, b, Y, M, N, K,
                                                          accum, relu);
  };
  auto ZERO = [&](float* p, long long n) {
    fill_val<<<nblk(n, 256), 256, 0, stream>>>(p, 0.0f, n);
  };
  auto AGG = [&](const float* h, const int* s, const int* d, int E, int F) {
    long long tot = (long long)E * (F >> 2);
    edge_gather_add<<<nblk(tot, 256), 256, 0, stream>>>(h, s, d, agg, cnt, E, F);
  };
  auto MEAN = [&](int n, int F) {
    long long tot = (long long)n * F;
    mean_div<<<nblk(tot, 256), 256, 0, stream>>>(agg, cnt, tot, F);
  };

  // ---- input projections: h = relu(x @ W + b) ----
  GEMM(x_emp, lin_emp_w, lin_emp_b, h_emp, NE_, H_, DEMP_, 0, 1);
  GEMM(x_per, lin_per_w, lin_per_b, h_per, NPER_, H_, DPER_, 0, 1);

  // ---- conv1: SAGE ee (emp->emp) ----
  ZERO(agg, (long long)NE_ * H_); ZERO(cnt, NE_);
  AGG(h_emp, ee_src, ee_dst, EEE_, H_); MEAN(NE_, H_);
  GEMM(agg,   s1_ee_Wl, s1_ee_bl, emp1, NE_, H_, H_, 0, 0);
  GEMM(h_emp, s1_ee_Wr, nullptr,  emp1, NE_, H_, H_, 1, 0);
  // ---- conv1: SAGE pe (per->emp), summed into emp1, relu on last ----
  ZERO(agg, (long long)NE_ * H_); ZERO(cnt, NE_);
  AGG(h_per, pe_src, pe_dst, EPE_, H_); MEAN(NE_, H_);
  GEMM(agg,   s1_pe_Wl, s1_pe_bl, emp1, NE_, H_, H_, 1, 0);
  GEMM(h_emp, s1_pe_Wr, nullptr,  emp1, NE_, H_, H_, 1, 1);
  // ---- conv1: SAGE ep (emp->per), relu ----
  ZERO(agg, (long long)NPER_ * H_); ZERO(cnt, NPER_);
  AGG(h_emp, ep_src, ep_dst, EEP_, H_); MEAN(NPER_, H_);
  GEMM(agg,   s1_ep_Wl, s1_ep_bl, per1, NPER_, H_, H_, 0, 0);
  GEMM(h_per, s1_ep_Wr, nullptr,  per1, NPER_, H_, H_, 1, 1);

  // ---- conv2: GAT (emp->emp) into out_emp ----
  GEMM(emp1, gat_Wsrc, nullptr, zs, NE_, OUT_, H_, 0, 0);
  GEMM(emp1, gat_Wdst, nullptr, zd, NE_, OUT_, H_, 0, 0);
  matvec_dot<<<nblk(NE_, 256), 256, 0, stream>>>(zs, gat_asrc, es, NE_, OUT_);
  matvec_dot<<<nblk(NE_, 256), 256, 0, stream>>>(zd, gat_adst, ed, NE_, OUT_);
  fill_val<<<nblk(NE_, 256), 256, 0, stream>>>(mmax, -3.0e38f, NE_);
  ZERO(den, NE_);
  gat_logit_max<<<nblk(EEE_, 256), 256, 0, stream>>>(es, ed, ee_src, ee_dst,
                                                     logit, mmax, EEE_);
  gat_exp_den<<<nblk(EEE_, 256), 256, 0, stream>>>(logit, mmax, ee_dst, den, EEE_);
  fill_bias_bcast<<<nblk((long long)NE_ * OUT_, 256), 256, 0, stream>>>(
      out_emp, gat_b, (long long)NE_ * OUT_, OUT_);
  {
    long long tot = (long long)EEE_ * (OUT_ >> 2);
    gat_scatter<<<nblk(tot, 256), 256, 0, stream>>>(logit, den, zs, ee_src,
                                                    ee_dst, out_emp, EEE_, OUT_);
  }
  // ---- conv2: SAGE pe (per1->emp), summed into out_emp ----
  ZERO(agg, (long long)NE_ * H_); ZERO(cnt, NE_);
  AGG(per1, pe_src, pe_dst, EPE_, H_); MEAN(NE_, H_);
  GEMM(agg,  s2_pe_Wl, s2_pe_bl, out_emp, NE_, OUT_, H_, 1, 0);
  GEMM(emp1, s2_pe_Wr, nullptr,  out_emp, NE_, OUT_, H_, 1, 0);
  // ---- conv2: SAGE ep (emp1->per) into out_per ----
  ZERO(agg, (long long)NPER_ * H_); ZERO(cnt, NPER_);
  AGG(emp1, ep_src, ep_dst, EEP_, H_); MEAN(NPER_, H_);
  GEMM(agg,  s2_ep_Wl, s2_ep_bl, out_per, NPER_, OUT_, H_, 0, 0);
  GEMM(per1, s2_ep_Wr, nullptr,  out_per, NPER_, OUT_, H_, 1, 0);
}
